// Model_37151467110551
// MI455X (gfx1250) — matrix operand order repaired and checked by reading, not compiled, not run on hardware
//
#include <hip/hip_runtime.h>

// ---------------------------------------------------------------------------
// Types / helpers
// ---------------------------------------------------------------------------
typedef __attribute__((ext_vector_type(16))) __bf16       v16bf;
typedef __attribute__((ext_vector_type(8)))  float        v8f;
typedef __attribute__((ext_vector_type(4)))  unsigned int u32x4;
typedef __attribute__((ext_vector_type(4)))  int          int32x4;
typedef __attribute__((ext_vector_type(8)))  int          int32x8;

#define WMMA_BF16(a, b, c) \
  __builtin_amdgcn_wmma_f32_16x16x32_bf16(false, (a), false, (b), (short)0, (c), false, false)

__device__ __forceinline__ v8f v8f_zero() {
  v8f v = {0.f, 0.f, 0.f, 0.f, 0.f, 0.f, 0.f, 0.f};
  return v;
}

// Build a 32-byte WMMA operand fragment from two 16-byte loads.
__device__ __forceinline__ v16bf ld_frag(const __bf16* a, const __bf16* b) {
  union { u32x4 q[2]; v16bf v; } u;
  u.q[0] = *(const u32x4*)a;
  u.q[1] = *(const u32x4*)b;
  return u.v;
}

__device__ __forceinline__ float redmax16(float v) {
#pragma unroll
  for (int off = 1; off < 16; off <<= 1) v = fmaxf(v, __shfl_xor(v, off, 32));
  return v;
}
__device__ __forceinline__ float redsum16(float v) {
#pragma unroll
  for (int off = 1; off < 16; off <<= 1) v += __shfl_xor(v, off, 32);
  return v;
}

// Async DMA of one 16-byte chunk global -> LDS (VGPR-bypassing, ASYNCcnt).
__device__ __forceinline__ void async_b128(const __bf16* gsrc, __bf16* ldst) {
  unsigned lds_off = (unsigned)(size_t)ldst;            // low 32b = wave LDS offset
  unsigned long long ga = (unsigned long long)(size_t)gsrc;
  asm volatile("global_load_async_to_lds_b128 %0, %1, off"
               :: "v"(lds_off), "v"(ga) : "memory");
}

// ---------------------------------------------------------------------------
// Tensor Data Mover: 2D tile (tile_w_dw x tile_h DWORDs) global -> LDS with
// engine-applied LDS padding of 4 DWORDs after every 16 DWORDs (row pitch
// 16+4 DW = 40 bf16, matching the padded LDS layout). D# per ISA ch.8.
// This toolchain exposes the 6-arg builtin:
//   (uint32x4 g0, int32x8 g1, int32x4 g2, int32x4 g3, int32x8 g4, i32 cpol)
// ---------------------------------------------------------------------------
#if __has_builtin(__builtin_amdgcn_tensor_load_to_lds)
#define HAVE_TDM 1
__device__ __forceinline__ void tdm_load_2d(const void* gaddr, unsigned lds_off,
                                            unsigned row_dw,   // tensor_dim0 & stride (DW)
                                            unsigned nrows,    // tensor_dim1
                                            unsigned tile_w_dw, unsigned tile_h) {
  unsigned long long ga = (unsigned long long)(size_t)gaddr;
  u32x4 g0;
  g0.x = 1u;                                 // count=1, user mode, no gather
  g0.y = lds_off;                            // lds_addr (bytes)
  g0.z = (unsigned)ga;                       // global_addr[31:0]
  g0.w = (unsigned)(ga >> 32) | (2u << 30);  // global_addr[56:32] | type=2
  int32x8 g1;
  g1[0] = (int)((2u << 16) | (1u << 20) | (3u << 22) | (3u << 25));
  //            data_size=4B  pad_enable  pad_int=16DW  pad_amt=4DW
  g1[1] = (int)((row_dw & 0xFFFFu) << 16);                      // tensor_dim0 lo
  g1[2] = (int)((row_dw >> 16) | ((nrows & 0xFFFFu) << 16));    // dim0 hi | dim1 lo
  g1[3] = (int)((nrows >> 16) | (tile_w_dw << 16));             // dim1 hi | tile_dim0
  g1[4] = (int)tile_h;                                          // tile_dim1 (tile_dim2=0)
  g1[5] = (int)row_dw;                                          // tensor_dim0_stride lo
  g1[6] = 0;                                                    // stride hi | dim1_stride
  g1[7] = 0;
  int32x4 z4 = {0, 0, 0, 0};
  int32x8 z8 = {0, 0, 0, 0, 0, 0, 0, 0};
  __builtin_amdgcn_tensor_load_to_lds(g0, g1, z4, z4, z8, 0);
}
#else
#define HAVE_TDM 0
#endif

// ---------------------------------------------------------------------------
// fp32 -> bf16 convert (weights)
// ---------------------------------------------------------------------------
__global__ __launch_bounds__(256) void cvt_kernel(const float* __restrict__ in,
                                                  __bf16* __restrict__ out, int n) {
  int i = blockIdx.x * 256 + threadIdx.x;
  if (i < n) out[i] = (__bf16)in[i];
}

// ---------------------------------------------------------------------------
// LayerNorm over D=1024, fp32 in -> bf16 out. One block per row.
// ---------------------------------------------------------------------------
__global__ __launch_bounds__(256) void ln_kernel(const float* __restrict__ x,
                                                 const float* __restrict__ g,
                                                 const float* __restrict__ bta,
                                                 __bf16* __restrict__ out) {
  const int D = 1024;
  int row = blockIdx.x;
  int t = threadIdx.x;
  const float* xr = x + (size_t)row * D;
  float v0 = xr[t], v1 = xr[t + 256], v2 = xr[t + 512], v3 = xr[t + 768];
  __shared__ float red[256];
  red[t] = v0 + v1 + v2 + v3;
  __syncthreads();
  for (int off = 128; off > 0; off >>= 1) {
    if (t < off) red[t] += red[t + off];
    __syncthreads();
  }
  float mu = red[0] * (1.0f / 1024.0f);
  __syncthreads();
  float d0 = v0 - mu, d1 = v1 - mu, d2 = v2 - mu, d3 = v3 - mu;
  red[t] = d0 * d0 + d1 * d1 + d2 * d2 + d3 * d3;
  __syncthreads();
  for (int off = 128; off > 0; off >>= 1) {
    if (t < off) red[t] += red[t + off];
    __syncthreads();
  }
  float rs = rsqrtf(red[0] * (1.0f / 1024.0f) + 1e-5f);
  __bf16* orow = out + (size_t)row * D;
  orow[t]       = (__bf16)(d0 * rs * g[t]       + bta[t]);
  orow[t + 256] = (__bf16)(d1 * rs * g[t + 256] + bta[t + 256]);
  orow[t + 512] = (__bf16)(d2 * rs * g[t + 512] + bta[t + 512]);
  orow[t + 768] = (__bf16)(d3 * rs * g[t + 768] + bta[t + 768]);
}

// ---------------------------------------------------------------------------
// WMMA GEMM: C[M,N] = A[M,K] * W[N,K]^T + bias (+res) (relu), A/W bf16.
// Block = 256 thr (8 waves), block tile 128x128, wave tile 32x64, K-step 32.
// A tile streams via TDM tensor_load_to_lds (wave 0, TENSORcnt), B tile via
// per-thread global_load_async_to_lds_b128 (ASYNCcnt). Double-buffered LDS.
// ---------------------------------------------------------------------------
template <bool RELU, bool RES, bool BF16OUT>
__global__ __launch_bounds__(256) void gemm_kernel(
    const __bf16* __restrict__ A, const __bf16* __restrict__ W,
    const float* __restrict__ bias, const float* __restrict__ res,
    float* __restrict__ outf, __bf16* __restrict__ outb, int M, int N, int K) {
  __shared__ alignas(16) __bf16 lds_a[2][128][40];
  __shared__ alignas(16) __bf16 lds_b[2][128][40];
  int n0 = blockIdx.x * 128, m0 = blockIdx.y * 128;
  int tid = threadIdx.x;
  int wave = tid >> 5, lane = tid & 31;
  int lh = lane & 15, sel = lane >> 4;
  int wm = (wave >> 1) * 32, wn = (wave & 1) * 64;

  // Per-thread B-tile copy assignment: 512 16B chunks per 128x32 operand tile.
  int row0 = tid >> 2,         kc0 = (tid & 3) * 8;
  int row1 = (tid + 256) >> 2, kc1 = kc0;
  const __bf16* gb0 = W + (size_t)(n0 + row0) * K + kc0;
  const __bf16* gb1 = W + (size_t)(n0 + row1) * K + kc1;
#if !HAVE_TDM
  const __bf16* ga0 = A + (size_t)(m0 + row0) * K + kc0;
  const __bf16* ga1 = A + (size_t)(m0 + row1) * K + kc1;
#endif

  v8f acc[2][4];
#pragma unroll
  for (int i = 0; i < 2; ++i)
#pragma unroll
    for (int j = 0; j < 4; ++j) acc[i][j] = v8f_zero();

  // Prologue: tile 0 -> buffer 0.
  async_b128(gb0, &lds_b[0][row0][kc0]);
  async_b128(gb1, &lds_b[0][row1][kc1]);
#if HAVE_TDM
  if (wave == 0)
    tdm_load_2d(A + (size_t)m0 * K, (unsigned)(size_t)&lds_a[0][0][0],
                (unsigned)(K >> 1), (unsigned)M, 16u, 128u);
#else
  async_b128(ga0, &lds_a[0][row0][kc0]);
  async_b128(ga1, &lds_a[0][row1][kc1]);
#endif

  int nk = K >> 5;
  for (int i = 0; i < nk; ++i) {
    int buf = i & 1;
    if (i + 1 < nk) {
      int ko = (i + 1) * 32;
      async_b128(gb0 + ko, &lds_b[buf ^ 1][row0][kc0]);
      async_b128(gb1 + ko, &lds_b[buf ^ 1][row1][kc1]);
#if HAVE_TDM
      if (wave == 0)
        tdm_load_2d(A + (size_t)m0 * K + ko, (unsigned)(size_t)&lds_a[buf ^ 1][0][0],
                    (unsigned)(K >> 1), (unsigned)M, 16u, 128u);
      asm volatile("s_wait_asynccnt 0x2" ::: "memory");  // tile i's B landed
      __builtin_amdgcn_s_wait_tensorcnt(1);              // tile i's A landed
#else
      async_b128(ga0 + ko, &lds_a[buf ^ 1][row0][kc0]);
      async_b128(ga1 + ko, &lds_a[buf ^ 1][row1][kc1]);
      asm volatile("s_wait_asynccnt 0x4" ::: "memory");
#endif
    } else {
      asm volatile("s_wait_asynccnt 0x0" ::: "memory");
#if HAVE_TDM
      __builtin_amdgcn_s_wait_tensorcnt(0);
#endif
    }
    __syncthreads();  // publish tile i across waves

    v16bf af[2], bf_[4];
#pragma unroll
    for (int sm = 0; sm < 2; ++sm) {     // A layout: K {0..7,16..23}/{8..15,24..31}
      int r = wm + sm * 16 + lh;
      af[sm] = ld_frag(&lds_a[buf][r][sel * 8], &lds_a[buf][r][16 + sel * 8]);
    }
#pragma unroll
    for (int sn = 0; sn < 4; ++sn) {     // B layout: lane=col, K {0..7,16..23}/{8..15,24..31}
      int r = wn + sn * 16 + lh;
      bf_[sn] = ld_frag(&lds_b[buf][r][sel * 8], &lds_b[buf][r][16 + sel * 8]);
    }
#pragma unroll
    for (int sm = 0; sm < 2; ++sm)
#pragma unroll
      for (int sn = 0; sn < 4; ++sn)
        acc[sm][sn] = WMMA_BF16(af[sm], bf_[sn], acc[sm][sn]);

    __syncthreads();  // all reads of buf done before tile i+2 streams into it
  }

#pragma unroll
  for (int sm = 0; sm < 2; ++sm)
#pragma unroll
    for (int sn = 0; sn < 4; ++sn)
#pragma unroll
      for (int r = 0; r < 8; ++r) {      // C layout: vgpr r -> row r (+8 for hi half)
        int m = m0 + wm + sm * 16 + r + sel * 8;
        int n = n0 + wn + sn * 16 + lh;
        float v = acc[sm][sn][r] + bias[n];
        if (RES) v += res[(size_t)m * N + n];
        if (RELU) v = fmaxf(v, 0.f);
        if (BF16OUT) outb[(size_t)m * N + n] = (__bf16)v;
        else         outf[(size_t)m * N + n] = v;
      }
}

// ---------------------------------------------------------------------------
// RoPE + QKV split. proj bf16 [S,B,3D] -> q,k bf16 [B,H,S,HD] (rotated),
// v transposed to [B,H,HD,S] so the PV WMMA B-operand is contiguous.
// ---------------------------------------------------------------------------
__global__ __launch_bounds__(256) void rope_kernel(
    const __bf16* __restrict__ proj, const float* __restrict__ sinp,
    const float* __restrict__ cosp, __bf16* __restrict__ q,
    __bf16* __restrict__ k, __bf16* __restrict__ vt) {
  int idx = blockIdx.x * 256 + threadIdx.x;  // S*B*H*(HD/2) = 2097152
  int i = idx & 31;
  int h = (idx >> 5) & 15;
  int b = (idx >> 9) & 1;
  int s = idx >> 10;
  size_t pbase = (size_t)(s * 2 + b) * 3072;
  int f = h * 64 + 2 * i;
  float sn = sinp[s * 32 + i], cs = cosp[s * 32 + i];
  float qe = (float)proj[pbase + f],        qo = (float)proj[pbase + f + 1];
  float ke = (float)proj[pbase + 1024 + f], ko = (float)proj[pbase + 1024 + f + 1];
  size_t qb = ((size_t)((b * 16 + h) * 2048 + s)) * 64 + 2 * i;
  q[qb]     = (__bf16)(qe * cs - qo * sn);
  q[qb + 1] = (__bf16)(qo * cs + qe * sn);
  k[qb]     = (__bf16)(ke * cs - ko * sn);
  k[qb + 1] = (__bf16)(ko * cs + ke * sn);
  size_t vb = ((size_t)(b * 16 + h) * 64 + 2 * i) * 2048 + s;
  vt[vb]        = proj[pbase + 2048 + f];
  vt[vb + 2048] = proj[pbase + 2048 + f + 1];
}

// ---------------------------------------------------------------------------
// Flash attention, causal. One wave per 16 query rows; WMMA for QK^T and PV.
// ---------------------------------------------------------------------------
__device__ __forceinline__ v8f score16(const __bf16* kp, int kbase, int q0,
                                       int lh, int sel, v16bf aq0, v16bf aq1) {
  const __bf16* kr = kp + (size_t)(kbase + lh) * 64;   // B operand: lane = key col
  v16bf bk0 = ld_frag(kr + sel * 8,      kr + 16 + sel * 8);       // hd 0..31
  v16bf bk1 = ld_frag(kr + 32 + sel * 8, kr + 48 + sel * 8);       // hd 32..63
  v8f s = v8f_zero();
  s = WMMA_BF16(aq0, bk0, s);
  s = WMMA_BF16(aq1, bk1, s);
  bool diag = (kbase == q0);
#pragma unroll
  for (int r = 0; r < 8; ++r) {
    float v = s[r] * 0.125f;                           // 1/sqrt(HD)
    if (diag && (kbase + lh > q0 + r + sel * 8)) v = -1e30f;
    s[r] = v;
  }
  return s;
}

__global__ __launch_bounds__(256) void attn_kernel(
    const __bf16* __restrict__ Q, const __bf16* __restrict__ Kb,
    const __bf16* __restrict__ VT, __bf16* __restrict__ O) {
  __shared__ alignas(16) __bf16 pbuf[8][16][40];  // per-wave P tile (16 x 32)
  int bh = blockIdx.x;                 // B*H = 32
  int b = bh >> 4, h = bh & 15;
  int wave = threadIdx.x >> 5, lane = threadIdx.x & 31;
  int lh = lane & 15, sel = lane >> 4;
  int q0 = (blockIdx.y * 8 + wave) * 16;

  const __bf16* qp = Q  + (size_t)bh * 2048 * 64;
  const __bf16* kp = Kb + (size_t)bh * 2048 * 64;
  const __bf16* vp = VT + (size_t)bh * 64 * 2048;

  const __bf16* qr = qp + (size_t)(q0 + lh) * 64;      // A operand: lane = query row
  v16bf aq0 = ld_frag(qr + sel * 8,      qr + 16 + sel * 8);
  v16bf aq1 = ld_frag(qr + 32 + sel * 8, qr + 48 + sel * 8);

  float mi[8], li[8];
  v8f oacc[4];
#pragma unroll
  for (int r = 0; r < 8; ++r) { mi[r] = -1e30f; li[r] = 0.f; }
#pragma unroll
  for (int n = 0; n < 4; ++n) oacc[n] = v8f_zero();

  int nblocks = (q0 >> 4) + 1;
  for (int jb0 = 0; jb0 < nblocks; jb0 += 2) {
    // Prefetch next pair's K rows (speculative; dropped if OOB).
    __builtin_prefetch(kp + (size_t)(jb0 * 16 + 32 + lh) * 64, 0, 0);

    v8f s0 = score16(kp, jb0 * 16, q0, lh, sel, aq0, aq1);
    v8f s1 = v8f_zero();
    bool has2 = (jb0 + 1) < nblocks;
    if (has2) {
      s1 = score16(kp, jb0 * 16 + 16, q0, lh, sel, aq0, aq1);
    } else {
#pragma unroll
      for (int r = 0; r < 8; ++r) s1[r] = -1e30f;
    }
    // online softmax (row r lives in vgpr r across a 16-lane half -> shfl reductions)
#pragma unroll
    for (int r = 0; r < 8; ++r) {
      float rm = redmax16(fmaxf(s0[r], s1[r]));
      float mnew = fmaxf(mi[r], rm);
      float alpha = __expf(mi[r] - mnew);
      float p0 = __expf(s0[r] - mnew);
      float p1 = __expf(s1[r] - mnew);
      li[r] = li[r] * alpha + redsum16(p0 + p1);
      mi[r] = mnew;
#pragma unroll
      for (int n = 0; n < 4; ++n) oacc[n][r] *= alpha;
      pbuf[wave][r + sel * 8][lh]      = (__bf16)p0;   // C layout -> LDS
      pbuf[wave][r + sel * 8][16 + lh] = (__bf16)p1;
    }
    asm volatile("s_wait_dscnt 0" ::: "memory");       // DS RAW before A-layout reads
    const __bf16* pr = &pbuf[wave][lh][0];
    v16bf ap = ld_frag(pr + sel * 8, pr + 16 + sel * 8);
    int kb32 = jb0 * 16;
#pragma unroll
    for (int n = 0; n < 4; ++n) {                      // B operand: lane = hd col, V^T rows
      const __bf16* vr = vp + (size_t)(n * 16 + lh) * 2048 + kb32;
      v16bf bv = ld_frag(vr + sel * 8, vr + 16 + sel * 8);
      oacc[n] = WMMA_BF16(ap, bv, oacc[n]);
    }
  }
#pragma unroll
  for (int n = 0; n < 4; ++n)
#pragma unroll
    for (int r = 0; r < 8; ++r) {
      float val = oacc[n][r] / li[r];
      int srow = q0 + r + sel * 8;
      O[((size_t)srow * 2 + b) * 1024 + h * 64 + n * 16 + lh] = (__bf16)val;
    }
}

// ---------------------------------------------------------------------------
// Host orchestration
// ---------------------------------------------------------------------------
extern "C" void kernel_launch(void* const* d_in, const int* in_sizes, int n_in,
                              void* d_out, int out_size, void* d_ws, size_t ws_size,
                              hipStream_t stream) {
  (void)in_sizes; (void)n_in; (void)out_size; (void)ws_size;
  const float* src   = (const float*)d_in[0];
  const float* sinp  = (const float*)d_in[1];
  const float* cosp  = (const float*)d_in[2];
  const float* in_w  = (const float*)d_in[3];
  const float* in_b  = (const float*)d_in[4];
  const float* out_w = (const float*)d_in[5];
  const float* out_b = (const float*)d_in[6];
  const float* ln1g  = (const float*)d_in[7];
  const float* ln1b  = (const float*)d_in[8];
  const float* ln2g  = (const float*)d_in[9];
  const float* ln2b  = (const float*)d_in[10];
  const float* w1    = (const float*)d_in[11];
  const float* b1    = (const float*)d_in[12];
  const float* w2    = (const float*)d_in[13];
  const float* b2    = (const float*)d_in[14];

  char* p = (char*)d_ws;
  auto take = [&](size_t elems, size_t esz) {
    void* r = (void*)p;
    p += (elems * esz + 255) & ~(size_t)255;
    return r;
  };
  __bf16* xln  = (__bf16*)take((size_t)4096 * 1024, 2);   // ln1(src)
  __bf16* wqkv = (__bf16*)take((size_t)3072 * 1024, 2);
  __bf16* wob  = (__bf16*)take((size_t)1024 * 1024, 2);
  __bf16* w1b  = (__bf16*)take((size_t)4096 * 1024, 2);
  __bf16* w2b  = (__bf16*)take((size_t)1024 * 4096, 2);
  __bf16* proj = (__bf16*)take((size_t)4096 * 3072, 2);   // qkv projection
  __bf16* qb   = (__bf16*)take((size_t)4194304, 2);       // [B,H,S,HD]
  __bf16* kb   = (__bf16*)take((size_t)4194304, 2);
  __bf16* vtb  = (__bf16*)take((size_t)4194304, 2);       // [B,H,HD,S]
  __bf16* ob   = (__bf16*)take((size_t)4096 * 1024, 2);   // attn out [S,B,D]
  float*  xres = (float*) take((size_t)4096 * 1024, 4);   // residual x
  __bf16* h2   = (__bf16*)take((size_t)4096 * 1024, 2);   // ln2(x)
  __bf16* h1   = (__bf16*)take((size_t)4096 * 4096, 2);   // relu(ffn1)

  // weights -> bf16
  cvt_kernel<<<3072 * 1024 / 256, 256, 0, stream>>>(in_w,  wqkv, 3072 * 1024);
  cvt_kernel<<<1024 * 1024 / 256, 256, 0, stream>>>(out_w, wob,  1024 * 1024);
  cvt_kernel<<<4096 * 1024 / 256, 256, 0, stream>>>(w1,    w1b,  4096 * 1024);
  cvt_kernel<<<4096 * 1024 / 256, 256, 0, stream>>>(w2,    w2b,  4096 * 1024);

  // ln1 -> qkv gemm -> rope/split -> attention -> out proj (+src residual)
  ln_kernel<<<4096, 256, 0, stream>>>(src, ln1g, ln1b, xln);
  gemm_kernel<false, false, true><<<dim3(24, 32), 256, 0, stream>>>(
      xln, wqkv, in_b, nullptr, nullptr, proj, 4096, 3072, 1024);
  rope_kernel<<<8192, 256, 0, stream>>>(proj, sinp, cosp, qb, kb, vtb);
  attn_kernel<<<dim3(32, 16), 256, 0, stream>>>(qb, kb, vtb, ob);
  gemm_kernel<false, true, false><<<dim3(8, 32), 256, 0, stream>>>(
      ob, wob, out_b, src, xres, nullptr, 4096, 1024, 1024);

  // ln2 -> ffn1 (relu) -> ffn2 (+x residual) -> d_out
  ln_kernel<<<4096, 256, 0, stream>>>(xres, ln2g, ln2b, h2);
  gemm_kernel<true, false, true><<<dim3(32, 32), 256, 0, stream>>>(
      h2, w1b, b1, nullptr, nullptr, h1, 4096, 4096, 1024);
  gemm_kernel<false, true, false><<<dim3(8, 32), 256, 0, stream>>>(
      h1, w2b, b2, xres, (float*)d_out, nullptr, 4096, 1024, 4096);
}
